// GatedMambaBlock_40329742909525
// MI455X (gfx1250) — matrix operand order repaired and checked by reading, not compiled, not run on hardware
//
#include <hip/hip_runtime.h>

#define BQ     4
#define LQ     2048
#define DIMQ   256
#define DSTATE 16
#define DINNER 512
#define DTRANK 16
#define NTOK   (BQ * LQ)          // 8192 tokens

typedef __attribute__((ext_vector_type(16))) __bf16 v16bf;
typedef __attribute__((ext_vector_type(8)))  __bf16 v8bf;
typedef __attribute__((ext_vector_type(8)))  float  v8f;
typedef __attribute__((ext_vector_type(4)))  float  v4f;
typedef unsigned int u32x4 __attribute__((ext_vector_type(4)));
typedef int          i32x8 __attribute__((ext_vector_type(8)));
typedef int          i32x4 __attribute__((ext_vector_type(4)));

static __device__ __forceinline__ __bf16 f2bf(float f) { return (__bf16)f; }
static __device__ __forceinline__ float sigm(float x) { return 1.f / (1.f + __expf(-x)); }

// ------------------------------------------------------------ TDM tile load
// 2D tile load via Tensor Data Mover: 128 rows x 32 bf16 elements, row stride
// = K elements, from global to LDS. D# packed per ISA 8.3/8.4:
//   group0: count=1 | lds_addr | global_addr[56:0] | type=2
//   group1: data_size=2B, tensor_dim0=32, tensor_dim1=rows_avail (OOB rows
//           read as zero -> handles the N=48 tail), tile 32x128, dim0 stride=K
// clang-23 toolchain: 6-arg builtin (g0, g1, g2, g3, g4, cpol).
static __device__ __forceinline__ void tdm_load_2d(unsigned lds_off,
                                                   const __bf16* gptr,
                                                   int rows_avail,
                                                   int row_stride_elems) {
  unsigned long long ga = (unsigned long long)(size_t)gptr;
  u32x4 g0;
  g0[0] = 1u;                                            // count=1, user mode
  g0[1] = lds_off;                                       // LDS byte address
  g0[2] = (unsigned)ga;                                  // global addr lo
  g0[3] = ((unsigned)(ga >> 32) & 0x1FFFFFFu) | (2u << 30);  // hi25 | type=2
  i32x8 g1;
  g1[0] = (1 << 16);                                     // data_size=1 -> 2B
  g1[1] = (int)(32u << 16);                              // tensor_dim0=32 lo16
  g1[2] = (int)(((unsigned)rows_avail & 0xFFFFu) << 16); // tensor_dim1 lo16
  g1[3] = (int)((((unsigned)rows_avail >> 16) & 0xFFFFu) | (32u << 16)); // dim1 hi | tile_dim0=32
  g1[4] = 128;                                           // tile_dim1=128, tile_dim2=0
  g1[5] = row_stride_elems;                              // tensor_dim0_stride lo32
  g1[6] = 0;                                             // stride hi16, dim1_stride lo
  g1[7] = 0;
  i32x4 z4 = {0, 0, 0, 0};
  i32x8 z8 = {0, 0, 0, 0, 0, 0, 0, 0};
#if defined(__has_builtin) && __has_builtin(__builtin_amdgcn_tensor_load_to_lds)
  __builtin_amdgcn_tensor_load_to_lds(g0, g1, z4, z4, z8, 0);
#endif
}

// ---------------------------------------------------------------- weight cvt
__global__ void k_f32_to_bf16(const float* __restrict__ src,
                              __bf16* __restrict__ dst, int n) {
  int i = blockIdx.x * blockDim.x + threadIdx.x;
  if (i < n) dst[i] = f2bf(src[i]);
}

// ---------------------------------------------------------------- layernorm
__global__ void k_layernorm_bf16(const float* __restrict__ x,
                                 const float* __restrict__ g,
                                 const float* __restrict__ b,
                                 __bf16* __restrict__ xn) {
  int wave = threadIdx.x >> 5;
  int lane = threadIdx.x & 31;
  int tok  = blockIdx.x * (blockDim.x >> 5) + wave;
  if (tok >= NTOK) return;
  const float* row = x + (size_t)tok * DIMQ + lane * 8;
  v4f a = *(const v4f*)row;
  v4f c = *(const v4f*)(row + 4);
  float s = a.x + a.y + a.z + a.w + c.x + c.y + c.z + c.w;
  for (int m = 16; m >= 1; m >>= 1) s += __shfl_xor(s, m, 32);
  float mu = s * (1.f / DIMQ);
  float q = 0.f;
#pragma unroll
  for (int i = 0; i < 4; i++) { float t = a[i] - mu; q += t * t; }
#pragma unroll
  for (int i = 0; i < 4; i++) { float t = c[i] - mu; q += t * t; }
  for (int m = 16; m >= 1; m >>= 1) q += __shfl_xor(q, m, 32);
  float rstd = rsqrtf(q * (1.f / DIMQ) + 1e-5f);
  __bf16* orow = xn + (size_t)tok * DIMQ + lane * 8;
  const float* gp = g + lane * 8;
  const float* bp = b + lane * 8;
#pragma unroll
  for (int i = 0; i < 8; i++) {
    float v = (i < 4) ? a[i] : c[i - 4];
    orow[i] = f2bf((v - mu) * rstd * gp[i] + bp[i]);
  }
}

// ---------------------------------------------------------------- WMMA GEMM
// C[M,N] = A[M,K] * W[N,K]^T, bf16 in / f32 accumulate.
// Block = 256 thr = 8 waves; block tile 128M x 128N; wave tile 32M x 64N
// (2x4 grid of 16x16 WMMA tiles, 8 v8f accumulators).
// K stepped by 32; A/B 128x32 tiles double-buffered in LDS, filled by the
// Tensor Data Mover (wave 0 issues, TENSORcnt-tracked, barrier-published).
__global__ void __launch_bounds__(256)
k_gemm_bf16(const __bf16* __restrict__ A, const __bf16* __restrict__ W,
            float* __restrict__ C, int M, int N, int K) {
  __shared__ __bf16 Abuf[2][128 * 32];
  __shared__ __bf16 Bbuf[2][128 * 32];

  int tid  = threadIdx.x;
  int wave = tid >> 5, lane = tid & 31;
  int half = lane >> 4, r = lane & 15;
  int wm   = wave >> 1;                 // 0..3 -> M offset wm*32
  int wn   = wave & 1;                  // 0..1 -> N offset wn*64
  int bm   = blockIdx.x * 128;
  int bn   = blockIdx.y * 128;

  unsigned aoff[2] = { (unsigned)(size_t)&Abuf[0][0], (unsigned)(size_t)&Abuf[1][0] };
  unsigned boff[2] = { (unsigned)(size_t)&Bbuf[0][0], (unsigned)(size_t)&Bbuf[1][0] };
  int arows = M - bm;                   // always >= 128 here
  int brows = N - bn;                   // 48 for the x_proj tail -> zero-fill

  v8f zero = {};
  v8f acc[2][4];
#pragma unroll
  for (int mt = 0; mt < 2; ++mt)
#pragma unroll
    for (int nt = 0; nt < 4; ++nt) acc[mt][nt] = zero;

  int nk = K / 32;
  bool w0 = (wave == 0);
  if (w0) {
    tdm_load_2d(aoff[0], A + (size_t)bm * K, arows, K);
    tdm_load_2d(boff[0], W + (size_t)bn * K, brows, K);
  }

  for (int ks = 0; ks < nk; ++ks) {
    int cur = ks & 1;
    __syncthreads();                    // all waves done reading buf cur^1
    if (w0) {
      if (ks + 1 < nk) {
        tdm_load_2d(aoff[cur ^ 1], A + (size_t)bm * K + (size_t)(ks + 1) * 32, arows, K);
        tdm_load_2d(boff[cur ^ 1], W + (size_t)bn * K + (size_t)(ks + 1) * 32, brows, K);
#if defined(__has_builtin) && __has_builtin(__builtin_amdgcn_s_wait_tensorcnt)
        __builtin_amdgcn_s_wait_tensorcnt(2);   // cur's 2 loads retired
#endif
      } else {
#if defined(__has_builtin) && __has_builtin(__builtin_amdgcn_s_wait_tensorcnt)
        __builtin_amdgcn_s_wait_tensorcnt(0);
#endif
      }
    }
    __syncthreads();                    // publish LDS tiles to all waves

    const __bf16* Ab = &Abuf[cur][0];
    const __bf16* Bb = &Bbuf[cur][0];

    v16bf afrag[2];
#pragma unroll
    for (int mt = 0; mt < 2; ++mt) {
      const __bf16* arow = Ab + (wm * 32 + mt * 16 + r) * 32;
      afrag[mt] = *(const v16bf*)(arow + half * 16);
    }
#pragma unroll
    for (int nt = 0; nt < 4; ++nt) {
      const __bf16* brow = Bb + (wn * 64 + nt * 16 + r) * 32 + half * 16;
      v16bf bfrag = *(const v16bf*)brow;
#pragma unroll
      for (int mt = 0; mt < 2; ++mt)
        acc[mt][nt] = __builtin_amdgcn_wmma_f32_16x16x32_bf16(
            false, afrag[mt], false, bfrag, (short)0, acc[mt][nt], false, false);
    }
  }

#pragma unroll
  for (int mt = 0; mt < 2; ++mt)
#pragma unroll
    for (int nt = 0; nt < 4; ++nt) {
      int col = bn + wn * 64 + nt * 16 + r;
      if (col < N) {
        float* crow = C + (size_t)(bm + wm * 32 + mt * 16 + half * 8) * N + col;
#pragma unroll
        for (int i = 0; i < 8; i++) crow[(size_t)i * N] = acc[mt][nt][i];
      }
    }
}

// ---------------------------------------------------------------- conv+silu
__global__ void k_conv_silu(const float* __restrict__ xz,
                            const float* __restrict__ cw,   // (512,1,4)
                            const float* __restrict__ cb,
                            float* __restrict__ u,
                            __bf16* __restrict__ ubf) {
  int idx = blockIdx.x * blockDim.x + threadIdx.x;   // (b*L + l)*512 + d
  if (idx >= NTOK * DINNER) return;
  int d = idx & (DINNER - 1);
  int t = idx >> 9;
  int l = t & (LQ - 1);
  float acc = cb[d];
  const float* w = cw + d * 4;
#pragma unroll
  for (int j = 0; j < 4; j++) {
    int ll = l - 3 + j;
    if (ll >= 0) acc += w[j] * xz[(size_t)(t - 3 + j) * (2 * DINNER) + d];
  }
  float s = acc * sigm(acc);
  u[idx] = s;
  ubf[idx] = f2bf(s);
}

// ---------------------------------------------------------------- dt_proj
__global__ void k_delta(const float* __restrict__ xdbl,   // (NTOK,48)
                        const float* __restrict__ dtw,    // (512,16)
                        const float* __restrict__ dtb,
                        float* __restrict__ delta) {
  int idx = blockIdx.x * blockDim.x + threadIdx.x;
  if (idx >= NTOK * DINNER) return;
  int d = idx & (DINNER - 1);
  int t = idx >> 9;
  const float* row = xdbl + (size_t)t * 48;
  const float* w   = dtw + d * 16;
  float acc = dtb[d];
#pragma unroll
  for (int r = 0; r < 16; r++) acc += row[r] * w[r];
  delta[idx] = (acc > 20.f) ? acc : log1pf(__expf(acc));
}

// ---------------------------------------------------------------- scan
// 16 lanes per channel (lane = state), 2 channels/wave, 1024 waves total.
__global__ void k_scan(const float* __restrict__ delta,
                       const float* __restrict__ xdbl,
                       const float* __restrict__ u,
                       const float* __restrict__ xz,
                       const float* __restrict__ A_log,
                       const float* __restrict__ Dp,
                       __bf16* __restrict__ ybf) {
  int lane = threadIdx.x & 31;
  int s    = lane & 15;
  int grp  = threadIdx.x >> 4;
  int c    = blockIdx.x * 16 + grp;             // 0..2047 = b*512 + d
  int b    = c >> 9;
  int d    = c & (DINNER - 1);

  float Aval = -__expf(A_log[d * DSTATE + s]);
  float Dd   = Dp[d];
  float h = 0.f, ystash = 0.f;
  size_t tb = (size_t)b * LQ;

  for (int t = 0; t < LQ; ++t) {
    size_t tok = tb + t;
    float dt = delta[tok * DINNER + d];
    float ut = u[tok * DINNER + d];
    const float* xr = xdbl + tok * 48;
    float Bt = xr[16 + s];
    float Ct = xr[32 + s];
    float dA = __expf(dt * Aval);
    h = dA * h + dt * Bt * ut;
    float v = h * Ct;
    v += __shfl_xor(v, 1, 32);
    v += __shfl_xor(v, 2, 32);
    v += __shfl_xor(v, 4, 32);
    v += __shfl_xor(v, 8, 32);
    if (s == (t & 15)) ystash = v;
    if ((t & 15) == 15) {
      size_t tt = tb + (size_t)(t - 15 + s);
      float uu = u[tt * DINNER + d];
      float zz = xz[tt * (2 * DINNER) + DINNER + d];
      float yf = (ystash + Dd * uu) * (zz * sigm(zz));
      ybf[tt * DINNER + d] = f2bf(yf);
      // prefetch the next window's streamed operands into cache
      __builtin_prefetch(delta + (tok + 16) * DINNER + d, 0, 0);
      __builtin_prefetch(u + (tok + 16) * DINNER + d, 0, 0);
      __builtin_prefetch(xdbl + (tok + 16) * 48 + s, 0, 0);
    }
  }
}

// ---------------------------------------------------------------- final fuse
__global__ void k_final(const float* __restrict__ x,
                        const float* __restrict__ mamba,
                        const float* __restrict__ glin,
                        const float* __restrict__ gb,
                        float* __restrict__ out) {
  int idx = blockIdx.x * blockDim.x + threadIdx.x;
  if (idx >= NTOK * DIMQ) return;
  int e = idx & (DIMQ - 1);
  out[idx] = x[idx] + mamba[idx] * sigm(glin[idx] + gb[e]);
}

// ================================================================ launch
extern "C" void kernel_launch(void* const* d_in, const int* in_sizes, int n_in,
                              void* d_out, int out_size, void* d_ws, size_t ws_size,
                              hipStream_t stream) {
  const float* x     = (const float*)d_in[0];
  const float* lng   = (const float*)d_in[1];
  const float* lnb   = (const float*)d_in[2];
  const float* inW   = (const float*)d_in[3];   // (1024,256)
  const float* convW = (const float*)d_in[4];   // (512,1,4)
  const float* convB = (const float*)d_in[5];
  const float* xpW   = (const float*)d_in[6];   // (48,512)
  const float* dtW   = (const float*)d_in[7];   // (512,16)
  const float* dtB   = (const float*)d_in[8];
  const float* Alog  = (const float*)d_in[9];
  const float* Dp    = (const float*)d_in[10];
  const float* outW  = (const float*)d_in[11];  // (256,512)
  const float* gW    = (const float*)d_in[12];  // (256,256)
  const float* gB    = (const float*)d_in[13];
  float* out = (float*)d_out;

  char* p = (char*)d_ws;
  auto alloc = [&](size_t bytes) -> char* {
    char* r = p; p += (bytes + 255) & ~(size_t)255; return r;
  };
  __bf16* xn_bf  = (__bf16*)alloc((size_t)NTOK * DIMQ * 2);
  __bf16* winbf  = (__bf16*)alloc((size_t)1024 * 256 * 2);
  __bf16* wxpbf  = (__bf16*)alloc((size_t)48 * 512 * 2);
  __bf16* woutbf = (__bf16*)alloc((size_t)256 * 512 * 2);
  __bf16* wgbf   = (__bf16*)alloc((size_t)256 * 256 * 2);
  __bf16* u_bf   = (__bf16*)alloc((size_t)NTOK * DINNER * 2);
  __bf16* y_bf   = (__bf16*)alloc((size_t)NTOK * DINNER * 2);
  float* xz    = (float*)alloc((size_t)NTOK * 2 * DINNER * 4);
  float* u_f   = (float*)alloc((size_t)NTOK * DINNER * 4);
  float* xdbl  = (float*)alloc((size_t)NTOK * 48 * 4);
  float* delta = (float*)alloc((size_t)NTOK * DINNER * 4);
  float* mamba = (float*)alloc((size_t)NTOK * DIMQ * 4);
  float* glin  = (float*)alloc((size_t)NTOK * DIMQ * 4);

  auto cvt = [&](const float* src, __bf16* dst, int n) {
    k_f32_to_bf16<<<(n + 255) / 256, 256, 0, stream>>>(src, dst, n);
  };
  cvt(inW,  winbf,  1024 * 256);
  cvt(xpW,  wxpbf,  48 * 512);
  cvt(outW, woutbf, 256 * 512);
  cvt(gW,   wgbf,   256 * 256);

  k_layernorm_bf16<<<NTOK / 8, 256, 0, stream>>>(x, lng, lnb, xn_bf);

  // in_proj: (8192,256) x (1024,256)^T -> xz
  { dim3 g(NTOK / 128, 1024 / 128);
    k_gemm_bf16<<<g, 256, 0, stream>>>(xn_bf, winbf, xz, NTOK, 1024, 256); }

  k_conv_silu<<<(NTOK * DINNER) / 256, 256, 0, stream>>>(xz, convW, convB, u_f, u_bf);

  // x_proj: (8192,512) x (48,512)^T -> x_dbl (TDM zero-fills rows 48..127)
  { dim3 g(NTOK / 128, 1);
    k_gemm_bf16<<<g, 256, 0, stream>>>(u_bf, wxpbf, xdbl, NTOK, 48, 512); }

  k_delta<<<(NTOK * DINNER) / 256, 256, 0, stream>>>(xdbl, dtW, dtB, delta);

  k_scan<<<(BQ * DINNER) / 16, 256, 0, stream>>>(delta, xdbl, u_f, xz, Alog, Dp, y_bf);

  // out_proj: (8192,512) x (256,512)^T -> mamba_out
  { dim3 g(NTOK / 128, 256 / 128);
    k_gemm_bf16<<<g, 256, 0, stream>>>(y_bf, woutbf, mamba, NTOK, 256, 512); }

  // gate GEMM: (8192,256) x (256,256)^T -> glin
  { dim3 g(NTOK / 128, 256 / 128);
    k_gemm_bf16<<<g, 256, 0, stream>>>(xn_bf, wgbf, glin, NTOK, 256, 256); }

  k_final<<<(NTOK * DIMQ) / 256, 256, 0, stream>>>(x, mamba, glin, gB, out);

  (void)in_sizes; (void)n_in; (void)out_size; (void)ws_size;
}
